// LlamaAttention_89867895701679
// MI455X (gfx1250) — compile-verified, matrix operand order checked by reading, not run on hardware
//
#include <hip/hip_runtime.h>

typedef __bf16 bf16;
typedef __attribute__((ext_vector_type(16))) __bf16 v16bf;
typedef __attribute__((ext_vector_type(8)))  __bf16 v8bf;
typedef __attribute__((ext_vector_type(8)))  float  v8f;
typedef __attribute__((ext_vector_type(4)))  int    v4i;

#define HID   2048
#define SEQ   2048
#define BATCH 2
#define NHEAD 16
#define HDIM  128
#define MROWS (BATCH*SEQ)   /* 4096 */

// ---- CDNA5 async global->LDS path (guarded; sync fallback keeps compile safe) ----
#if defined(__has_builtin)
#if __has_builtin(__builtin_amdgcn_global_load_async_to_lds_b128) && \
    __has_builtin(__builtin_amdgcn_s_wait_asynccnt)
#define USE_ASYNC_LDS 1
#endif
#endif
#ifndef USE_ASYNC_LDS
#define USE_ASYNC_LDS 0
#endif

// Stage 16 bytes global -> LDS.
__device__ __forceinline__ void stage16(const bf16* g, bf16* l) {
#if USE_ASYNC_LDS
  __builtin_amdgcn_global_load_async_to_lds_b128(
      (__attribute__((address_space(1))) v4i*)g,
      (__attribute__((address_space(3))) v4i*)l, 0, 0);
#else
  *reinterpret_cast<v8bf*>(l) = *reinterpret_cast<const v8bf*>(g);
#endif
}

__device__ __forceinline__ void wait_async_lds() {
#if USE_ASYNC_LDS
  __builtin_amdgcn_s_wait_asynccnt(0);
#endif
}

// ---------- fragment helpers ----------

__device__ __forceinline__ v16bf join8(v8bf lo, v8bf hi) {
  return __builtin_shufflevector(lo, hi, 0,1,2,3,4,5,6,7,8,9,10,11,12,13,14,15);
}

// WMMA k=32 fragment from a row-major row (ISA 7.12.2 16-bit A/B pattern):
// e<8 -> k=8*half+e ; e>=8 -> k=16+8*half+(e-8)
__device__ __forceinline__ v16bf frag_k32(const bf16* rowk0, int half) {
  v8bf lo = *reinterpret_cast<const v8bf*>(rowk0 + 8*half);
  v8bf hi = *reinterpret_cast<const v8bf*>(rowk0 + 16 + 8*half);
  return join8(lo, hi);
}

__device__ __forceinline__ float rmax16(float v) {
  #pragma unroll
  for (int m = 1; m < 16; m <<= 1) v = fmaxf(v, __shfl_xor(v, m, 16));
  return v;
}
__device__ __forceinline__ float rsum16(float v) {
  #pragma unroll
  for (int m = 1; m < 16; m <<= 1) v += __shfl_xor(v, m, 16);
  return v;
}

__device__ __forceinline__ v8f wmma_bf16(v16bf a, v16bf b, v8f c) {
  return __builtin_amdgcn_wmma_f32_16x16x32_bf16(false, a, false, b, (short)0, c, false, false);
}

// ---------- fp32 -> bf16 conversion ----------

__global__ __launch_bounds__(256) void cvt_f32_bf16(const float* __restrict__ in,
                                                    bf16* __restrict__ out, int n) {
  int i = blockIdx.x * blockDim.x + threadIdx.x;
  if (i < n) out[i] = (bf16)in[i];
}

// ---------- GEMM: out[m,n] = sum_k A[m,k] * W[n,k]  (x @ W^T) ----------
// 256 threads / 8 waves; block tile 128x128; K stepped by 32 with LDS double
// buffering (async global->LDS). Wave computes 64x32 via 4 A-frags x 2 B-frags
// = 8 WMMAs per k-step.

enum { OUT_BF16_ROW = 0, OUT_F32_ROW = 1, OUT_BF16_VT = 2 };

template<int OUTMODE>
__global__ __launch_bounds__(256) void gemm_xwT(const bf16* __restrict__ A,
                                                const bf16* __restrict__ W,
                                                void* __restrict__ outp,
                                                int M, int N, int K) {
  __shared__ __align__(16) bf16 As[2][128][40];   // 32 valid + 8 pad
  __shared__ __align__(16) bf16 Bs[2][128][40];

  const int tid  = threadIdx.x;
  const int lane = tid & 31;
  const int w    = tid >> 5;
  const int half = lane >> 4;
  const int lrow = lane & 15;
  const int wm   = w >> 2;          // 0..1 -> 64-row quadrant
  const int wn   = w & 3;           // 0..3 -> 32-col quadrant

  const int ntiles = N >> 7;
  const int mt = blockIdx.x / ntiles;
  const int nt = blockIdx.x - mt * ntiles;
  const int m0 = mt * 128, n0 = nt * 128;

  // staging: 2 threads per row, each moves 32B (two b128)
  const int srow = tid >> 1;                 // 0..127
  const int soff = (tid & 1) * 16;           // elements
  const bf16* gA = A + (size_t)(m0 + srow) * K + soff;
  const bf16* gB = W + (size_t)(n0 + srow) * K + soff;

  auto stage_tiles = [&](int buf, int k0) {
    stage16(gA + k0,     &As[buf][srow][soff]);
    stage16(gA + k0 + 8, &As[buf][srow][soff + 8]);
    stage16(gB + k0,     &Bs[buf][srow][soff]);
    stage16(gB + k0 + 8, &Bs[buf][srow][soff + 8]);
  };

  v8f acc[4][2] = {};
  const int nsteps = K >> 5;

  stage_tiles(0, 0);
  wait_async_lds();
  __syncthreads();

  for (int it = 0; it < nsteps; ++it) {
    const int buf = it & 1;
    if (it + 1 < nsteps) stage_tiles(buf ^ 1, (it + 1) * 32);   // overlap with math

    v16bf b0 = frag_k32(&Bs[buf][wn * 32 + lrow][0], half);
    v16bf b1 = frag_k32(&Bs[buf][wn * 32 + 16 + lrow][0], half);
    #pragma unroll
    for (int f = 0; f < 4; ++f) {
      v16bf af = frag_k32(&As[buf][wm * 64 + f * 16 + lrow][0], half);
      acc[f][0] = wmma_bf16(af, b0, acc[f][0]);
      acc[f][1] = wmma_bf16(af, b1, acc[f][1]);
    }

    wait_async_lds();
    __syncthreads();
  }

  // C/D layout: VGPR r, lanes 0-15 -> M=r ; lanes 16-31 -> M=8+r ; N = lane%16
  #pragma unroll
  for (int f = 0; f < 4; ++f) {
    #pragma unroll
    for (int g = 0; g < 2; ++g) {
      #pragma unroll
      for (int r = 0; r < 8; ++r) {
        const int row = m0 + wm * 64 + f * 16 + r + 8 * half;
        const int col = n0 + wn * 32 + g * 16 + lrow;
        const float val = acc[f][g][r];
        if constexpr (OUTMODE == OUT_BF16_ROW) {
          ((bf16*)outp)[(size_t)row * N + col] = (bf16)val;
        } else if constexpr (OUTMODE == OUT_F32_ROW) {
          ((float*)outp)[(size_t)row * N + col] = val;
        } else {  // OUT_BF16_VT: V transposed to [B][H][D][S] (keys contiguous)
          const int h = col >> 7, d = col & (HDIM - 1);
          const int b = row >> 11, s = row & (SEQ - 1);
          ((bf16*)outp)[((size_t)(b * NHEAD + h) * HDIM + d) * SEQ + s] = (bf16)val;
        }
      }
    }
  }
}

// ---------- Flash attention: one wave per (b, h, 16-query tile) ----------
// Q/K: bf16 [B*S, HID] rows.  V: bf16 transposed [B][H][D][S].

__global__ __launch_bounds__(256) void attn_flash(const bf16* __restrict__ Q,
                                                  const bf16* __restrict__ Kc,
                                                  const bf16* __restrict__ Vt,
                                                  bf16* __restrict__ O) {
  __shared__ __align__(16) bf16 plds[8][16][32];   // per-wave P tile (16 q x 32 keys)

  const int lane = threadIdx.x & 31;
  const int w    = threadIdx.x >> 5;
  const int half = lane >> 4;
  const int lrow = lane & 15;

  const int qtiles = SEQ / 16;                     // 128
  const int wid = blockIdx.x * 8 + w;
  const int qt  = wid % qtiles;
  const int bh  = wid / qtiles;
  const int h   = bh % NHEAD;
  const int b   = bh / NHEAD;

  const int    q0      = qt * 16;
  const size_t rowbase = (size_t)b * SEQ;
  const int    hoff    = h * HDIM;
  const bf16*  vtb     = Vt + (size_t)(b * NHEAD + h) * HDIM * SEQ;

  // Resident Q fragments: 16 rows x 128 d, as 4 k=32 A-fragments
  v16bf qf[4];
  {
    const bf16* qrow = Q + (rowbase + q0 + lrow) * HID + hoff;
    #pragma unroll
    for (int c = 0; c < 4; ++c) qf[c] = frag_k32(qrow + c * 32, half);
  }

  v8f o[8] = {};                 // 16 q x 128 d accumulator (8 C tiles)
  float mrow[8], lsum[8];
  #pragma unroll
  for (int r = 0; r < 8; ++r) { mrow[r] = -3.0e38f; lsum[r] = 0.0f; }

  const float scale = 0.08838834764831845f;        // 1/sqrt(128)
  const int nkt = (q0 + 15) / 32 + 1;              // causal bound, 32-key tiles

  for (int kt = 0; kt < nkt; ++kt) {
    const int kb = kt * 32;

    // scores: S = Q . K^T  (contract over d; two 16-key groups)
    v8f s0 = {}, s1 = {};
    const bf16* kr0 = Kc + (rowbase + kb + lrow) * HID + hoff;
    const bf16* kr1 = Kc + (rowbase + kb + 16 + lrow) * HID + hoff;
    #pragma unroll
    for (int c = 0; c < 4; ++c) {
      v16bf b0 = frag_k32(kr0 + c * 32, half);
      v16bf b1 = frag_k32(kr1 + c * 32, half);
      s0 = wmma_bf16(qf[c], b0, s0);
      s1 = wmma_bf16(qf[c], b1, s1);
    }

    // online softmax (one row = one VGPR across 16 lanes)
    #pragma unroll
    for (int r = 0; r < 8; ++r) {
      const int qi = q0 + r + 8 * half;
      float v0 = s0[r] * scale; if (kb + lrow > qi)      v0 = -3.0e38f;
      float v1 = s1[r] * scale; if (kb + 16 + lrow > qi) v1 = -3.0e38f;
      float mx   = rmax16(fmaxf(v0, v1));
      float mnew = fmaxf(mrow[r], mx);
      float alpha = __expf(mrow[r] - mnew);
      mrow[r] = mnew;
      float p0 = __expf(v0 - mnew);
      float p1 = __expf(v1 - mnew);
      lsum[r] = lsum[r] * alpha + rsum16(p0 + p1);
      #pragma unroll
      for (int c = 0; c < 8; ++c) o[c][r] *= alpha;
      const int m = r + 8 * half;
      plds[w][m][lrow]      = (bf16)p0;
      plds[w][m][16 + lrow] = (bf16)p1;
    }

    // P in A-layout from LDS (same-wave DS dependency)
    v16bf pf = join8(*reinterpret_cast<const v8bf*>(&plds[w][lrow][8 * half]),
                     *reinterpret_cast<const v8bf*>(&plds[w][lrow][16 + 8 * half]));

    // O += P . V : contraction over 32 keys; V^T rows are key-contiguous
    #pragma unroll
    for (int c = 0; c < 8; ++c) {
      v16bf vf = frag_k32(vtb + (size_t)(c * 16 + lrow) * SEQ + kb, half);
      o[c] = wmma_bf16(pf, vf, o[c]);
    }
  }

  // epilogue: O /= l, write bf16 attn-out in [B*S, HID] layout
  #pragma unroll
  for (int r = 0; r < 8; ++r) {
    const float inv = 1.0f / lsum[r];
    const int row   = q0 + r + 8 * half;
    const size_t ob = (rowbase + row) * HID + hoff + lrow;
    #pragma unroll
    for (int c = 0; c < 8; ++c) O[ob + c * 16] = (bf16)(o[c][r] * inv);
  }
}

// ---------- launch ----------

extern "C" void kernel_launch(void* const* d_in, const int* in_sizes, int n_in,
                              void* d_out, int out_size, void* d_ws, size_t ws_size,
                              hipStream_t stream) {
  (void)in_sizes; (void)n_in; (void)out_size; (void)ws_size;
  const float* hs = (const float*)d_in[0];
  const float* wq = (const float*)d_in[1];
  const float* wk = (const float*)d_in[2];
  const float* wv = (const float*)d_in[3];
  const float* wo = (const float*)d_in[4];

  char* ws = (char*)d_ws;
  const size_t XB = (size_t)MROWS * HID * 2;   // 16 MB
  const size_t WB = (size_t)HID * HID * 2;     //  8 MB
  bf16* Xb  = (bf16*)(ws);
  bf16* Wqb = (bf16*)(ws + XB);
  bf16* Wkb = (bf16*)(ws + XB + WB);
  bf16* Wvb = (bf16*)(ws + XB + 2 * WB);
  bf16* Wob = (bf16*)(ws + XB + 3 * WB);
  bf16* Qb  = (bf16*)(ws + XB + 4 * WB);
  bf16* Kb  = (bf16*)(ws + 2 * XB + 4 * WB);
  bf16* Vtb = (bf16*)(ws + 3 * XB + 4 * WB);
  bf16* Ab  = (bf16*)(ws + 4 * XB + 4 * WB);

  // 1) fp32 -> bf16
  const int nX = MROWS * HID, nW = HID * HID;
  cvt_f32_bf16<<<(nX + 255) / 256, 256, 0, stream>>>(hs, Xb, nX);
  cvt_f32_bf16<<<(nW + 255) / 256, 256, 0, stream>>>(wq, Wqb, nW);
  cvt_f32_bf16<<<(nW + 255) / 256, 256, 0, stream>>>(wk, Wkb, nW);
  cvt_f32_bf16<<<(nW + 255) / 256, 256, 0, stream>>>(wv, Wvb, nW);
  cvt_f32_bf16<<<(nW + 255) / 256, 256, 0, stream>>>(wo, Wob, nW);

  // 2) projections: (4096/128)*(2048/128) = 512 blocks of 8 waves
  const int gemmBlocks = (MROWS / 128) * (HID / 128);
  gemm_xwT<OUT_BF16_ROW><<<gemmBlocks, 256, 0, stream>>>(Xb, Wqb, Qb,  MROWS, HID, HID);
  gemm_xwT<OUT_BF16_ROW><<<gemmBlocks, 256, 0, stream>>>(Xb, Wkb, Kb,  MROWS, HID, HID);
  gemm_xwT<OUT_BF16_VT ><<<gemmBlocks, 256, 0, stream>>>(Xb, Wvb, Vtb, MROWS, HID, HID);

  // 3) flash attention: B*H*(S/16) = 4096 waves / 8
  attn_flash<<<(BATCH * NHEAD * (SEQ / 16)) / 8, 256, 0, stream>>>(Qb, Kb, Vtb, Ab);

  // 4) output projection (fp32 out)
  gemm_xwT<OUT_F32_ROW><<<gemmBlocks, 256, 0, stream>>>(Ab, Wob, (float*)d_out, MROWS, HID, HID);
}
